// CoordinatedDREAMStack_14267881357952
// MI455X (gfx1250) — compile-verified, matrix operand order checked by reading, not run on hardware
//
#include <hip/hip_runtime.h>
#include <hip/hip_bf16.h>
#include <math.h>

// Problem dims (match reference)
#define BB 64
#define TT 512
#define DD 512
#define HH 1024
#define RR 16

typedef __attribute__((ext_vector_type(16))) __bf16 v16bf;
typedef __attribute__((ext_vector_type(8)))  __bf16 v8bf;
typedef __attribute__((ext_vector_type(8)))  float  v8f;

// ---------------------------------------------------------------------------
// bf16 fragment loader (wave32, 16x16x32 bf16 WMMA layouts, CDNA5 ISA 7.12.2)
//
// A (16x32, MxK): lane m (0-15) holds row m with K = kb+0..7 and kb+16..23;
//                 lane m+16 holds row m with K = kb+8..15 and kb+24..31.
// B (32x16, KxN): weights stored transposed (N-major, ld = K), so each lane
//                 reads its column contiguously with the same K striping.
// `p` is the per-lane base (row/col picked by lane&15, +8 K-offset for the
// upper lane half already applied); two 16B loads per fragment, no VALU.
// ---------------------------------------------------------------------------
__device__ __forceinline__ v16bf load_frag(const __bf16* __restrict__ p) {
  v8bf lo = *(const v8bf*)p;          // K = kb+0..7   (or +8..15 in hi lanes)
  v8bf hi = *(const v8bf*)(p + 16);   // K = kb+16..23 (or +24..31)
  return __builtin_shufflevector(lo, hi, 0, 1, 2, 3, 4, 5, 6, 7,
                                         8, 9, 10, 11, 12, 13, 14, 15);
}

// ---------------------------------------------------------------------------
// DREAM cell:  h_new = tanh( X @ Wx  +  Hp @ Wr_eff  +  b )
// Wr_eff already contains the folded rank-R fast weights (V @ U^T).
// All GEMM operands are bf16; accumulation f32 in the WMMA accumulator.
// One wave -> one 16x16 tile; 8 waves/block cover 64(M) x 32(N); grid HH/32.
// Epilogue writes f32 state (for err-subtract / output) AND a bf16 copy that
// feeds the next GEMM's A-fragments directly (no per-iteration conversion).
// ---------------------------------------------------------------------------
__global__ __launch_bounds__(256)
void dream_cell(const __bf16* __restrict__ Xbf, long ldx, int Kx,
                const __bf16* __restrict__ Hbf,   // prev state, [BB x HH] bf16
                const __bf16* __restrict__ WxT,   // [HH x Kx] bf16, N-major
                const __bf16* __restrict__ WrT,   // [HH x HH] bf16, N-major
                const float* __restrict__ bias,
                float* __restrict__ Hn,           // new state, f32
                __bf16* __restrict__ Hnbf,        // new state, bf16
                float* __restrict__ Y, long ldy) {
  const int lane  = threadIdx.x & 31;
  const int wave  = threadIdx.x >> 5;
  const int mbase = (wave & 3) * 16;
  const int nbase = blockIdx.x * 32 + (wave >> 2) * 16;
  const int khalf = (lane >> 4) << 3;
  const int lrow  = lane & 15;

  v8f acc = {};

  // ---- X @ Wx ----
  {
    const __bf16* Ap = Xbf + (size_t)(mbase + lrow) * ldx + khalf;
    const __bf16* Bp = WxT + (size_t)(nbase + lrow) * Kx + khalf;
    for (int kb = 0; kb < Kx; kb += 32) {
      v16bf a = load_frag(Ap + kb);
      v16bf b = load_frag(Bp + kb);
      acc = __builtin_amdgcn_wmma_f32_16x16x32_bf16(false, a, false, b,
                                                    (short)0, acc, false, false);
    }
  }
  // ---- Hp @ Wr_eff ----
  {
    const __bf16* Ap = Hbf + (size_t)(mbase + lrow) * HH + khalf;
    const __bf16* Bp = WrT + (size_t)(nbase + lrow) * HH + khalf;
    for (int kb = 0; kb < HH; kb += 32) {
      v16bf a = load_frag(Ap + kb);
      v16bf b = load_frag(Bp + kb);
      acc = __builtin_amdgcn_wmma_f32_16x16x32_bf16(false, a, false, b,
                                                    (short)0, acc, false, false);
    }
  }

  // Epilogue: C/D layout — n = lane&15, rows m = (lane>=16 ? 8 : 0)+r
  const int n  = nbase + lrow;
  const int m0 = mbase + ((lane >> 4) << 3);
  const float bv = bias[n];
#pragma unroll
  for (int r = 0; r < 8; ++r) {
    float v = tanhf(acc[r] + bv);
    Hn  [(size_t)(m0 + r) * HH + n] = v;
    Hnbf[(size_t)(m0 + r) * HH + n] = (__bf16)v;
    if (Y) Y[(size_t)(m0 + r) * ldy + n] = v;
  }
}

// ---------------------------------------------------------------------------
// Final-step projections:
//   Sub == nullptr :  Out = sigmoid(Hbf @ W)        (mods)
//   Sub != nullptr :  Out = Hbf @ W - Sub           (errs, Sub is f32 state)
// ---------------------------------------------------------------------------
__global__ __launch_bounds__(256)
void dream_proj(const __bf16* __restrict__ Hbf,
                const __bf16* __restrict__ WT,   // [HH x HH] bf16, N-major
                const float* __restrict__ Sub,
                float* __restrict__ Out) {
  const int lane  = threadIdx.x & 31;
  const int wave  = threadIdx.x >> 5;
  const int mbase = (wave & 3) * 16;
  const int nbase = blockIdx.x * 32 + (wave >> 2) * 16;
  const int khalf = (lane >> 4) << 3;
  const int lrow  = lane & 15;

  v8f acc = {};
  const __bf16* Ap = Hbf + (size_t)(mbase + lrow) * HH + khalf;
  const __bf16* Bp = WT + (size_t)(nbase + lrow) * HH + khalf;
  for (int kb = 0; kb < HH; kb += 32) {
    v16bf a = load_frag(Ap + kb);
    v16bf b = load_frag(Bp + kb);
    acc = __builtin_amdgcn_wmma_f32_16x16x32_bf16(false, a, false, b,
                                                  (short)0, acc, false, false);
  }

  const int n  = nbase + lrow;
  const int m0 = mbase + ((lane >> 4) << 3);
#pragma unroll
  for (int r = 0; r < 8; ++r) {
    float v = acc[r];
    if (Sub) v = v - Sub[(size_t)(m0 + r) * HH + n];
    else     v = 1.0f / (1.0f + expf(-v));
    Out[(size_t)(m0 + r) * HH + n] = v;
  }
}

// ---------------------------------------------------------------------------
// Prep kernels
// ---------------------------------------------------------------------------
// transpose+convert f32 weight [K x N] -> bf16 [N x K]
__global__ void conv_transpose(const float* __restrict__ W,
                               __bf16* __restrict__ WT, int K, int N) {
  long i = (long)blockIdx.x * blockDim.x + threadIdx.x;
  if (i < (long)K * N) {
    int k = (int)(i / N);
    int n = (int)(i % N);
    WT[(size_t)n * K + k] = (__bf16)W[i];
  }
}

// WrT[n][k] = bf16( W_rec[k][n] + sum_r V[k][r] * U[n][r] )
// (folds the rank-R fast weights V @ U^T into the recurrent matrix)
__global__ void fold_fast(const float* __restrict__ Wrec,
                          const float* __restrict__ U,
                          const float* __restrict__ V,
                          __bf16* __restrict__ WT) {
  long i = (long)blockIdx.x * blockDim.x + threadIdx.x;
  if (i < (long)HH * HH) {
    int k = (int)(i / HH);
    int n = (int)(i % HH);
    float acc = Wrec[i];
#pragma unroll
    for (int r = 0; r < RR; ++r)
      acc += V[(size_t)k * RR + r] * U[(size_t)n * RR + r];
    WT[(size_t)n * HH + k] = (__bf16)acc;
  }
}

// elementwise f32 -> bf16 (same layout)
__global__ void conv_bf16(const float* __restrict__ A,
                          __bf16* __restrict__ O, long n) {
  long i = (long)blockIdx.x * blockDim.x + threadIdx.x;
  if (i < n) O[i] = (__bf16)A[i];
}

// per-timestep x slice [BB x DD] (row stride TT*DD) -> bf16 staging [BB x DD]
__global__ void conv_xslice(const float* __restrict__ Xt,
                            __bf16* __restrict__ O) {
  long i = (long)blockIdx.x * blockDim.x + threadIdx.x;
  if (i < (long)BB * DD) {
    long b = i / DD, d = i % DD;
    O[i] = (__bf16)Xt[b * (long)TT * DD + d];
  }
}

// zero f32 + bf16 state buffers
__global__ void zero_state(float* __restrict__ Hf, __bf16* __restrict__ Hb) {
  long i = (long)blockIdx.x * blockDim.x + threadIdx.x;
  if (i < (long)BB * HH) { Hf[i] = 0.0f; Hb[i] = (__bf16)0.0f; }
}

// ---------------------------------------------------------------------------
extern "C" void kernel_launch(void* const* d_in, const int* in_sizes, int n_in,
                              void* d_out, int out_size, void* d_ws, size_t ws_size,
                              hipStream_t stream) {
  (void)in_sizes; (void)n_in; (void)out_size;

  const float* x     = (const float*)d_in[0];
  const float* Win0  = (const float*)d_in[1];
  const float* Wrec0 = (const float*)d_in[2];
  const float* b0    = (const float*)d_in[3];
  const float* U0    = (const float*)d_in[4];
  const float* V0    = (const float*)d_in[5];
  const float* Win1  = (const float*)d_in[6];
  const float* Wrec1 = (const float*)d_in[7];
  const float* b1    = (const float*)d_in[8];
  const float* U1    = (const float*)d_in[9];
  const float* V1    = (const float*)d_in[10];
  const float* Win2  = (const float*)d_in[11];
  const float* Wrec2 = (const float*)d_in[12];
  const float* b2    = (const float*)d_in[13];
  const float* U2    = (const float*)d_in[14];
  const float* V2    = (const float*)d_in[15];
  const float* M1    = (const float*)d_in[16];
  const float* P1    = (const float*)d_in[17];
  const float* M2    = (const float*)d_in[18];
  const float* P2    = (const float*)d_in[19];

  float* out = (float*)d_out;

  // ---- workspace carve-out ----
  char* ws = (char*)d_ws;
  size_t off = 0;
  auto carve = [&](size_t bytes) -> void* {
    void* p = ws + off;
    off = (off + bytes + 255) & ~(size_t)255;
    return p;
  };
  float*  hf[3][2];
  __bf16* hb[3][2];
  for (int l = 0; l < 3; ++l)
    for (int j = 0; j < 2; ++j) {
      hf[l][j] = (float*) carve((size_t)BB * HH * sizeof(float));
      hb[l][j] = (__bf16*)carve((size_t)BB * HH * sizeof(__bf16));
    }
  __bf16* W0xT = (__bf16*)carve((size_t)HH * DD * sizeof(__bf16));
  __bf16* W1xT = (__bf16*)carve((size_t)HH * HH * sizeof(__bf16));
  __bf16* W2xT = (__bf16*)carve((size_t)HH * HH * sizeof(__bf16));
  __bf16* WrT0 = (__bf16*)carve((size_t)HH * HH * sizeof(__bf16));
  __bf16* WrT1 = (__bf16*)carve((size_t)HH * HH * sizeof(__bf16));
  __bf16* WrT2 = (__bf16*)carve((size_t)HH * HH * sizeof(__bf16));
  __bf16* M1T  = (__bf16*)carve((size_t)HH * HH * sizeof(__bf16));
  __bf16* P1T  = (__bf16*)carve((size_t)HH * HH * sizeof(__bf16));
  __bf16* M2T  = (__bf16*)carve((size_t)HH * HH * sizeof(__bf16));
  __bf16* P2T  = (__bf16*)carve((size_t)HH * HH * sizeof(__bf16));
  __bf16* xstage = (__bf16*)carve((size_t)BB * DD * sizeof(__bf16));

  // Whole-x bf16 conversion if workspace allows (33.5 MB); else per-step slices.
  const size_t xbytes = (size_t)BB * TT * DD * sizeof(__bf16);
  __bf16* xbf = nullptr;
  if (ws_size >= off + xbytes) xbf = (__bf16*)carve(xbytes);

  const int PT = 256;
  long nBH = (long)BB * HH, nDH = (long)DD * HH, nHH = (long)HH * HH;

  // ---- prep ----
  for (int l = 0; l < 3; ++l)
    for (int j = 0; j < 2; ++j)
      zero_state<<<dim3((nBH + PT - 1) / PT), dim3(PT), 0, stream>>>(hf[l][j], hb[l][j]);

  conv_transpose<<<dim3((nDH + PT - 1) / PT), dim3(PT), 0, stream>>>(Win0, W0xT, DD, HH);
  conv_transpose<<<dim3((nHH + PT - 1) / PT), dim3(PT), 0, stream>>>(Win1, W1xT, HH, HH);
  conv_transpose<<<dim3((nHH + PT - 1) / PT), dim3(PT), 0, stream>>>(Win2, W2xT, HH, HH);
  conv_transpose<<<dim3((nHH + PT - 1) / PT), dim3(PT), 0, stream>>>(M1, M1T, HH, HH);
  conv_transpose<<<dim3((nHH + PT - 1) / PT), dim3(PT), 0, stream>>>(P1, P1T, HH, HH);
  conv_transpose<<<dim3((nHH + PT - 1) / PT), dim3(PT), 0, stream>>>(M2, M2T, HH, HH);
  conv_transpose<<<dim3((nHH + PT - 1) / PT), dim3(PT), 0, stream>>>(P2, P2T, HH, HH);
  fold_fast<<<dim3((nHH + PT - 1) / PT), dim3(PT), 0, stream>>>(Wrec0, U0, V0, WrT0);
  fold_fast<<<dim3((nHH + PT - 1) / PT), dim3(PT), 0, stream>>>(Wrec1, U1, V1, WrT1);
  fold_fast<<<dim3((nHH + PT - 1) / PT), dim3(PT), 0, stream>>>(Wrec2, U2, V2, WrT2);
  if (xbf) {
    long nX = (long)BB * TT * DD;
    conv_bf16<<<dim3((nX + PT - 1) / PT), dim3(PT), 0, stream>>>(x, xbf, nX);
  }

  // ---- the scan: 3 dependent cell GEMMs per timestep ----
  dim3 grid(HH / 32), block(256);
  long nXS = (long)BB * DD;
  int cur = 0;
  for (int t = 0; t < TT; ++t) {
    int nxt = cur ^ 1;
    const __bf16* x_t;
    long ldx0;
    if (xbf) { x_t = xbf + (size_t)t * DD; ldx0 = (long)TT * DD; }
    else {
      conv_xslice<<<dim3((nXS + PT - 1) / PT), dim3(PT), 0, stream>>>(
          x + (size_t)t * DD, xstage);
      x_t = xstage; ldx0 = DD;
    }
    dream_cell<<<grid, block, 0, stream>>>(
        x_t, ldx0, DD,
        hb[0][cur], W0xT, WrT0, b0, hf[0][nxt], hb[0][nxt], nullptr, 0);
    dream_cell<<<grid, block, 0, stream>>>(
        hb[0][nxt], (long)HH, HH,
        hb[1][cur], W1xT, WrT1, b1, hf[1][nxt], hb[1][nxt], nullptr, 0);
    dream_cell<<<grid, block, 0, stream>>>(
        hb[1][nxt], (long)HH, HH,
        hb[2][cur], W2xT, WrT2, b2, hf[2][nxt], hb[2][nxt],
        out + (size_t)t * HH, (long)TT * HH);   // y[b][t][:] stream-out
    cur = nxt;
  }

  // ---- final-step mods / errs only (scan carry semantics) ----
  float* mods = out + (size_t)BB * TT * HH;
  float* errs = mods + (size_t)2 * BB * HH;
  dream_proj<<<grid, block, 0, stream>>>(hb[1][cur], M1T, nullptr, mods);
  dream_proj<<<grid, block, 0, stream>>>(hb[2][cur], M2T, nullptr, mods + (size_t)BB * HH);
  dream_proj<<<grid, block, 0, stream>>>(hb[1][cur], P1T, hf[0][cur], errs);
  dream_proj<<<grid, block, 0, stream>>>(hb[2][cur], P2T, hf[1][cur], errs + (size_t)BB * HH);
}
